// FineMatching_609885356632
// MI455X (gfx1250) — compile-verified, matrix operand order checked by reading, not run on hardware
//
#include <hip/hip_runtime.h>

typedef __attribute__((ext_vector_type(16))) _Float16 v16h;
typedef __attribute__((ext_vector_type(4)))  _Float16 v4h;
typedef __attribute__((ext_vector_type(8)))  float    v8f;
typedef __attribute__((ext_vector_type(4)))  float    v4f;

#define SAB_STRIDE 72   // halfs per staged row (64 data + 8 pad) -> conflict-free b128 frag loads
#define SC_STRIDE  113  // floats per conf row (112 cols + 1 pad) -> conflict-free col scans

// Load one 16x32 f16 WMMA operand fragment from LDS (row-major tile, stride SAB_STRIDE).
// ISA layout: lane = row%16 ; lanes 0-15 hold K = 0..7,16..23 ; lanes 16-31 hold K = 8..15,24..31.
__device__ __forceinline__ v16h load_frag(const _Float16* tile0, int lane, int kbase) {
    int row = lane & 15;
    int kh  = (lane >> 4) << 3;           // 0 or 8
    const _Float16* p = tile0 + row * SAB_STRIDE + kbase + kh;
    v16h f;
#pragma unroll
    for (int x = 0; x < 8; ++x) f[x] = p[x];
#pragma unroll
    for (int x = 0; x < 8; ++x) f[8 + x] = p[16 + x];
    return f;
}

__global__ __launch_bounds__(128)
void fine_matching_kernel(const float* __restrict__ feat0,   // (M,64,64)
                          const float* __restrict__ feat1,   // (M,100,64)
                          const float* __restrict__ mk0,     // (M,2)
                          const float* __restrict__ mk1,     // (M,2)
                          float* __restrict__ out,           // sm | mk0_f | mk1_f
                          int M)
{
    __shared__ _Float16 sA[64  * SAB_STRIDE];
    __shared__ _Float16 sB[112 * SAB_STRIDE];
    __shared__ float    sC[64 * SC_STRIDE];
    __shared__ float rowmax_s[64], rowsum_s[64];
    __shared__ float colmax_s[100], colsum_s[100];
    __shared__ float bestv_s[128];
    __shared__ int   besti_s[128];

    const int m    = blockIdx.x;
    const int tid  = threadIdx.x;
    const int lane = tid & 31;
    const int wv   = tid >> 5;      // wave id 0..3 = l-tile

    const float* f0 = feat0 + (size_t)m * 64 * 64;
    const float* f1 = feat1 + (size_t)m * 100 * 64;

    // ---- stage features to LDS as f16 (b128 global loads, b64 LDS stores) ----
    // pre-scaled by 1/sqrt(C)=1/8 for conf channels; ff channels (56..63) zeroed.
    {
        const v4f* f0v = (const v4f*)f0;              // 1024 float4
        for (int j = tid; j < 1024; j += 128) {
            int row = j >> 4, cb = (j & 15) << 2;     // cb multiple of 4; 56 is too
            v4f v = f0v[j];
            v4h h;
            if (cb < 56) {
#pragma unroll
                for (int x = 0; x < 4; ++x) h[x] = (_Float16)(v[x] * 0.125f);
            } else {
#pragma unroll
                for (int x = 0; x < 4; ++x) h[x] = (_Float16)0.0f;
            }
            *(v4h*)(sA + row * SAB_STRIDE + cb) = h;
        }
        const v4f* f1v = (const v4f*)f1;              // 1600 float4
        for (int j = tid; j < 1600; j += 128) {
            int row = j >> 4, cb = (j & 15) << 2;
            v4f v = f1v[j];
            v4h h;
            if (cb < 56) {
#pragma unroll
                for (int x = 0; x < 4; ++x) h[x] = (_Float16)(v[x] * 0.125f);
            } else {
#pragma unroll
                for (int x = 0; x < 4; ++x) h[x] = (_Float16)0.0f;
            }
            *(v4h*)(sB + row * SAB_STRIDE + cb) = h;
        }
        for (int j = tid; j < 192; j += 128) {        // zero pad rows 100..111
            int row = 100 + (j >> 4), cb = (j & 15) << 2;
            v4h h;
#pragma unroll
            for (int x = 0; x < 4; ++x) h[x] = (_Float16)0.0f;
            *(v4h*)(sB + row * SAB_STRIDE + cb) = h;
        }
    }
    __syncthreads();

    // ---- GEMM: conf(64x100) = A(64x56) * B(100x56)^T via v_wmma_f32_16x16x32_f16 ----
    {
        const _Float16* aT = sA + (16 * wv) * SAB_STRIDE;
        v16h a0 = load_frag(aT, lane, 0);
        v16h a1 = load_frag(aT, lane, 32);
        for (int rt = 0; rt < 7; ++rt) {
            const _Float16* bT = sB + (16 * rt) * SAB_STRIDE;
            v16h b0 = load_frag(bT, lane, 0);
            v16h b1 = load_frag(bT, lane, 32);
            v8f acc = {};
            acc = __builtin_amdgcn_wmma_f32_16x16x32_f16(false, a0, false, b0,
                                                         (short)0, acc, false, false);
            acc = __builtin_amdgcn_wmma_f32_16x16x32_f16(false, a1, false, b1,
                                                         (short)0, acc, false, false);
            // C/D layout: lane 0-15 -> N=lane, M=v ; lane 16-31 -> N=lane-16, M=v+8
            int ncol  = 16 * rt + (lane & 15);
            int mbase = 16 * wv + ((lane >> 4) << 3);
#pragma unroll
            for (int v = 0; v < 8; ++v)
                sC[(mbase + v) * SC_STRIDE + ncol] = acc[v];
        }
    }
    __syncthreads();

    // ---- softmax statistics: rows (axis=2 over r<100) and cols (axis=1 over l<64) ----
    if (tid < 64) {
        const float* row = sC + tid * SC_STRIDE;
        float mx = -3.0e38f;
        for (int r = 0; r < 100; ++r) mx = fmaxf(mx, row[r]);
        float s = 0.0f;
        for (int r = 0; r < 100; ++r) s += __expf(row[r] - mx);
        rowmax_s[tid] = mx; rowsum_s[tid] = s;
    }
    if (tid < 100) {
        float mx = -3.0e38f;
        for (int l = 0; l < 64; ++l) mx = fmaxf(mx, sC[l * SC_STRIDE + tid]);
        float s = 0.0f;
        for (int l = 0; l < 64; ++l) s += __expf(sC[l * SC_STRIDE + tid] - mx);
        colmax_s[tid] = mx; colsum_s[tid] = s;
    }
    __syncthreads();

    // ---- sm = softmax_l * softmax_r, cropped to interior 8x8 (float4 stores); fused argmax ----
    float bv = -3.0e38f; int bi = 0;
    float* outm = out + (size_t)m * 4096;
    for (int i4 = tid; i4 < 1024; i4 += 128) {
        int i  = i4 << 2;
        int l  = i >> 6;
        int j  = i & 63;                               // jj base is 0 or 4: stays inside one row
        int rb = ((j >> 3) + 1) * 10 + (j & 7) + 1;    // interior of 10x10
        float rm = rowmax_s[l], rs = rowsum_s[l];
        v4f o;
#pragma unroll
        for (int k = 0; k < 4; ++k) {
            float c = sC[l * SC_STRIDE + rb + k];
            float v = (__expf(c - rm) / rs) *
                      (__expf(c - colmax_s[rb + k]) / colsum_s[rb + k]);
            o[k] = v;
            if (v > bv) { bv = v; bi = i + k; }        // ascending index within thread
        }
        *(v4f*)(outm + i) = o;
    }
    bestv_s[tid] = bv; besti_s[tid] = bi;
    __syncthreads();

    // ---- epilogue: argmax reduce + keypoint refinement (single thread; tiny) ----
    if (tid == 0) {
        float gv = bestv_s[0]; int gi = besti_s[0];
        for (int t = 1; t < 128; ++t) {
            float v = bestv_s[t]; int ix = besti_s[t];
            if (v > gv || (v == gv && ix < gi)) { gv = v; gi = ix; }
        }
        int idx_l = gi >> 6, idx_r = gi & 63;
        size_t smtot = (size_t)M * 4096;

        // mkpts0_f = mkpts0_c + grid[idx_l] * SCALE,  grid[k] = (k/8-3.5, k%8-3.5)
        out[smtot + 2 * m + 0] = mk0[2 * m + 0] + ((float)(idx_l >> 3) - 3.5f) * 2.0f;
        out[smtot + 2 * m + 1] = mk0[2 * m + 1] + ((float)(idx_l & 7) - 3.5f) * 2.0f;

        int ir = idx_r >> 3, jr = idx_r & 7;
        float m1x = mk1[2 * m + 0] + ((float)ir - 3.5f) * 2.0f;
        float m1y = mk1[2 * m + 1] + ((float)jr - 3.5f) * 2.0f;

        // conf_ff only needed at the 3x3 window: win[p][q] = <ff0[idx_l], ff1[(ir+q)*10+(jr+p)]>/sqrt(8)
        float ffa[8];
#pragma unroll
        for (int c = 0; c < 8; ++c) ffa[c] = f0[idx_l * 64 + 56 + c];
        float w9[9];
        for (int p = 0; p < 3; ++p)
            for (int q = 0; q < 3; ++q) {
                int r2 = (ir + q) * 10 + (jr + p);
                float s = 0.0f;
#pragma unroll
                for (int c = 0; c < 8; ++c) s += ffa[c] * f1[r2 * 64 + 56 + c];
                w9[p * 3 + q] = s * 0.35355339059327373f;   // 1/sqrt(8)
            }
        // heat = softmax(w9 / TEMP)
        float mx = w9[0];
        for (int k = 1; k < 9; ++k) mx = fmaxf(mx, w9[k]);
        float hs = 0.0f, heat[9];
        for (int k = 0; k < 9; ++k) { heat[k] = __expf((w9[k] - mx) * 0.1f); hs += heat[k]; }
        for (int k = 0; k < 9; ++k) heat[k] /= hs;
        // 3x3 ones conv, zero pad; normalized centroid
        float cc[9], csum = 0.0f;
        for (int y = 0; y < 3; ++y)
            for (int x = 0; x < 3; ++x) {
                float s = 0.0f;
                for (int dy = -1; dy <= 1; ++dy)
                    for (int dx = -1; dx <= 1; ++dx) {
                        int yy = y + dy, xx = x + dx;
                        if (yy >= 0 && yy < 3 && xx >= 0 && xx < 3) s += heat[yy * 3 + xx];
                    }
                cc[y * 3 + x] = s; csum += s;
            }
        float cx = 0.0f, cy = 0.0f;
        for (int y = 0; y < 3; ++y)
            for (int x = 0; x < 3; ++x) {
                float w = cc[y * 3 + x] / csum;
                cx += w * (float)x; cy += w * (float)y;
            }
        out[smtot + (size_t)2 * M + 2 * m + 0] = m1x + (cx - 1.0f) * 2.0f;
        out[smtot + (size_t)2 * M + 2 * m + 1] = m1y + (cy - 1.0f) * 2.0f;
    }
}

extern "C" void kernel_launch(void* const* d_in, const int* in_sizes, int n_in,
                              void* d_out, int out_size, void* d_ws, size_t ws_size,
                              hipStream_t stream) {
    const float* feat0 = (const float*)d_in[0];
    const float* feat1 = (const float*)d_in[1];
    const float* mk0   = (const float*)d_in[2];
    const float* mk1   = (const float*)d_in[3];
    float* out = (float*)d_out;
    int M = in_sizes[2] / 2;   // mkpts0_c is (M,2)
    fine_matching_kernel<<<M, 128, 0, stream>>>(feat0, feat1, mk0, mk1, out, M);
}
